// Hypergraph_88630945120541
// MI455X (gfx1250) — matrix operand order repaired and checked by reading, not compiled, not run on hardware
//
#include <hip/hip_runtime.h>
#include <hip/hip_bf16.h>
#include <cmath>

#define NV_   100000
#define NE_   50000
#define NNZ_  500000
#define D_    128
#define NCLS_ 40
#define ALPHA_ 0.1f

typedef __attribute__((ext_vector_type(16))) __bf16 v16bf;
typedef __attribute__((ext_vector_type(8)))  float  v8f;

union FragAB { v16bf v; unsigned short h[16]; uint4 q[2]; };
union FragC  { v8f  v; float f[8]; };

__device__ __forceinline__ unsigned short f2bf(float x) {
  unsigned int u = __float_as_uint(x);
  u += 0x7fffu + ((u >> 16) & 1u);           // round-to-nearest-even
  return (unsigned short)(u >> 16);
}

// A fragment (16x32 bf16) from LDS, rows 0..15, cols k0..k0+31, row stride lda.
// ISA layout: lanes 0-15 hold M=lane, K=k0..k0+7 (v0-3) and k0+16..k0+23 (v4-7);
// lanes 16-31 hold K=k0+8.. and k0+24..  -> two contiguous 16B loads per lane.
__device__ __forceinline__ void loadA(FragAB& a, const unsigned short* as,
                                      int lda, int k0, int lane) {
  const int half = (lane >> 4) & 1, m = lane & 15;
  const unsigned short* p = as + m * lda + k0 + 8 * half;
  a.q[0] = *(const uint4*)(p);
  a.q[1] = *(const uint4*)(p + 16);
}

// B fragment (32x16 bf16) from pre-packed weights: tile-major, lane-major,
// 16 contiguous values per lane (u[j] = W[k0+16*(j>>3)+8*(lane>>4)+(j&7)][n0+(lane&15)]).
__device__ __forceinline__ void loadB(FragAB& b, const unsigned short* wpk,
                                      int nTiles, int k0, int n0, int lane) {
  const unsigned short* p =
      wpk + (((size_t)((k0 >> 5) * nTiles + (n0 >> 4))) << 9) + lane * 16;
  b.q[0] = *(const uint4*)(p);
  b.q[1] = *(const uint4*)(p + 8);
}

// ---------------------------------------------------------------- utility kernels

__global__ __launch_bounds__(256) void fill_zero_k(float* p, long long n) {
  long long t = (long long)blockIdx.x * blockDim.x + threadIdx.x;
  if (t < n) p[t] = 0.f;
}

__global__ __launch_bounds__(256) void seg_count_k(const int* idx, float* cnt, int n) {
  int t = blockIdx.x * blockDim.x + threadIdx.x;
  if (t < n) atomicAdd(&cnt[idx[t]], 1.f);
}

// Pack f32 weight [K][N] row-major into WMMA B-fragment layout, zero-padded to Npad cols.
__global__ __launch_bounds__(256) void pack_w_bf16_k(const float* w, unsigned short* dst,
                                                     int K, int N, int Npad) {
  const int total = (K >> 5) * (Npad >> 4) * 512;
  const int nT = Npad >> 4;
  for (int idx = blockIdx.x * blockDim.x + threadIdx.x; idx < total;
       idx += gridDim.x * blockDim.x) {
    int j    = idx & 15;
    int lane = (idx >> 4) & 31;
    int tile = idx >> 9;
    int nt = tile % nT, kt = tile / nT;
    int k = kt * 32 + ((j >> 3) << 4) + ((lane >> 4) << 3) + (j & 7);
    int n = nt * 16 + (lane & 15);
    float val = (n < N) ? w[(size_t)k * N + n] : 0.f;
    dst[idx] = f2bf(val);
  }
}

// acc[sidx[k]] += src[gidx[k]] * wrow[k]  (4 feats/thread via b128 load; wrow nullable)
__global__ __launch_bounds__(256) void gather_scatter_add_k(const float* src, const int* gidx,
                                                            const int* sidx, const float* wrow,
                                                            float* acc, int nnz) {
  long long t = (long long)blockIdx.x * blockDim.x + threadIdx.x;
  int k = (int)(t >> 5);
  if (k >= nnz) return;
  int d = (int)(t & 31) << 2;
  float s = wrow ? wrow[k] : 1.f;
  float4 v = *(const float4*)(src + (size_t)gidx[k] * D_ + d);
  float* ap = acc + (size_t)sidx[k] * D_ + d;
  atomicAdd(ap + 0, v.x * s);
  atomicAdd(ap + 1, v.y * s);
  atomicAdd(ap + 2, v.z * s);
  atomicAdd(ap + 3, v.w * s);
}

__global__ __launch_bounds__(256) void div_count_k(float* acc, const float* cnt, long long total) {
  long long t = (long long)blockIdx.x * blockDim.x + threadIdx.x;
  if (t < total) acc[t] /= fmaxf(cnt[t >> 7], 1.f);
}

// edge = ((1-a)*acc/max(cnt,1) + a*e0) / e_reg_sum   (in place on acc)
__global__ __launch_bounds__(256) void edge_finalize_k(float* edge, const float* cnt,
                                                       const float* e0, const float* ers,
                                                       long long total) {
  long long t = (long long)blockIdx.x * blockDim.x + threadIdx.x;
  if (t >= total) return;
  long long r = t >> 7;
  float mean = edge[t] / fmaxf(cnt[r], 1.f);
  edge[t] = ((1.f - ALPHA_) * mean + ALPHA_ * e0[t]) / ers[r];
}

// node = (1-a)*(acc/max(cnt,1))/n_reg_sum + a*v0
__global__ __launch_bounds__(256) void node_finalize_k(const float* acc, const float* cnt,
                                                       const float* nrs, const float* v0,
                                                       float* node, long long total) {
  long long t = (long long)blockIdx.x * blockDim.x + threadIdx.x;
  if (t >= total) return;
  long long r = t >> 7;
  float mean = acc[t] / fmaxf(cnt[r], 1.f);
  node[t] = (1.f - ALPHA_) * (mean / nrs[r]) + ALPHA_ * v0[t];
}

__global__ __launch_bounds__(256) void relu_copy_k(float* dst, const float* src, long long n) {
  long long t = (long long)blockIdx.x * blockDim.x + threadIdx.x;
  if (t < n) dst[t] = fmaxf(src[t], 0.f);
}

// ---------------------------------------------------------------- single linear + ReLU
// y[16 rows] = relu(x @ W + b), Din = Dout = 128. 8 waves, one 16x16 N-tile each.
__global__ __launch_bounds__(256) void linear_relu_k(const float* x, const unsigned short* wpk,
                                                     const float* bias, float* y) {
  __shared__ __attribute__((aligned(16))) unsigned short ab[16 * 128];
  const int tid = threadIdx.x;
  const size_t row0 = (size_t)blockIdx.x * 16;
  for (int idx = tid; idx < (16 * 128) / 4; idx += 256) {
    float4 v = *(const float4*)(x + row0 * 128 + idx * 4);
    ab[idx * 4 + 0] = f2bf(v.x);
    ab[idx * 4 + 1] = f2bf(v.y);
    ab[idx * 4 + 2] = f2bf(v.z);
    ab[idx * 4 + 3] = f2bf(v.w);
  }
  __syncthreads();
  const int lane = tid & 31, wave = tid >> 5;
  const int n0 = wave << 4;
  FragC c;
#pragma unroll
  for (int i = 0; i < 8; i++) c.f[i] = 0.f;
#pragma unroll
  for (int k0 = 0; k0 < 128; k0 += 32) {
    FragAB a, b;
    loadA(a, ab, 128, k0, lane);
    loadB(b, wpk, 8, k0, n0, lane);
    c.v = __builtin_amdgcn_wmma_f32_16x16x32_bf16(false, a.v, false, b.v,
                                                  (short)0, c.v, false, false);
  }
  const int half = lane >> 4, n = lane & 15;
  const float bs = bias[n0 + n];
#pragma unroll
  for (int i = 0; i < 8; i++) {
    size_t r = row0 + i + 8 * half;
    y[r * 128 + n0 + n] = fmaxf(c.f[i] + bs, 0.f);
  }
}

// ---------------------------------------------------------------- fused 2-layer MLP
// MODE 0: x = xsrc[row]*rowScale[row];                 out: plain store
// MODE 1: x = concat(xsrc[vidx[row]], xsrc2[eidx[row]]*rowScale[eidx[row]]);
//         out: atomicAdd(outp[vidx[row]], (beta*o+(1-beta)*blendSrc[eidx[row]])*rowW[row])
// MODE 2: x = xsrc[row]; out: relu(beta*o+(1-beta)*blendSrc[row])
// MODE 3: x = xsrc[row]; no input LN; out cols < DOUT stored (classifier)
template <int MODE, int DIN, int DHID, int DOUTPAD, int DOUT>
__global__ __launch_bounds__(256) void mlp_fused_k(
    const float* __restrict__ xsrc, const float* __restrict__ xsrc2,
    const int* __restrict__ gvidx, const int* __restrict__ geidx,
    const float* __restrict__ rowScale,
    const float* __restrict__ ln0g, const float* __restrict__ ln0b,
    const unsigned short* __restrict__ w0pk, const float* __restrict__ b0,
    const float* __restrict__ ln1g, const float* __restrict__ ln1b,
    const unsigned short* __restrict__ w1pk, const float* __restrict__ b1,
    float beta, const float* __restrict__ blendSrc, const float* __restrict__ rowW,
    float* __restrict__ outp) {
  __shared__ __attribute__((aligned(16))) float          xs[16 * DIN];
  __shared__ __attribute__((aligned(16))) unsigned short ab[16 * DIN];
  __shared__ __attribute__((aligned(16))) float          hs[16 * DHID];
  __shared__ __attribute__((aligned(16))) unsigned short hb[16 * DHID];
  __shared__ float red[16][16];
  __shared__ float red2[16][16];
  __shared__ float mrow[16], rrow[16];
  __shared__ int   ridxV[16], ridxE[16];

  const int tid  = threadIdx.x;
  const int row0 = blockIdx.x * 16;
  const int lane = tid & 31, wave = tid >> 5;

  if constexpr (MODE == 1) {
    if (tid < 16) {
      ridxV[tid] = gvidx[row0 + tid];
      ridxE[tid] = geidx[row0 + tid];
    }
  }
  __syncthreads();

  // ---- stage input rows into LDS (f32), vectorized b128
  constexpr int DIN4 = DIN / 4;
  for (int idx = tid; idx < 16 * DIN4; idx += 256) {
    int r = idx / DIN4, d = (idx - r * DIN4) * 4;
    float4 val;
    if constexpr (MODE == 1) {
      if (d < 128) {
        val = *(const float4*)(xsrc + (size_t)ridxV[r] * 128 + d);
      } else {
        val = *(const float4*)(xsrc2 + (size_t)ridxE[r] * 128 + (d - 128));
        float s = rowScale[ridxE[r]];
        val.x *= s; val.y *= s; val.z *= s; val.w *= s;
      }
    } else {
      val = *(const float4*)(xsrc + (size_t)(row0 + r) * DIN + d);
      if constexpr (MODE == 0) {
        float s = rowScale[row0 + r];
        val.x *= s; val.y *= s; val.z *= s; val.w *= s;
      }
    }
    *(float4*)(xs + idx * 4) = val;
  }
  __syncthreads();

  // ---- optional input LayerNorm, then bf16 into ab
  if constexpr (MODE != 3) {
    int r = tid >> 4, sub = tid & 15;
    float s = 0.f, s2 = 0.f;
    for (int d = sub; d < DIN; d += 16) { float w = xs[r * DIN + d]; s += w; s2 += w * w; }
    red[r][sub] = s; red2[r][sub] = s2;
    __syncthreads();
    if (sub == 0) {
      float ts = 0.f, ts2 = 0.f;
#pragma unroll
      for (int j = 0; j < 16; j++) { ts += red[r][j]; ts2 += red2[r][j]; }
      float m = ts / DIN;
      mrow[r] = m;
      rrow[r] = rsqrtf(ts2 / DIN - m * m + 1e-5f);
    }
    __syncthreads();
    for (int idx = tid; idx < 16 * DIN; idx += 256) {
      int r2 = idx / DIN, d = idx - r2 * DIN;
      ab[idx] = f2bf((xs[idx] - mrow[r2]) * rrow[r2] * ln0g[d] + ln0b[d]);
    }
  } else {
    for (int idx = tid; idx < 16 * DIN; idx += 256) ab[idx] = f2bf(xs[idx]);
  }
  __syncthreads();

  // ---- GEMM1: h = relu(x @ W0 + b0)  -> hs (f32 LDS)
  constexpr int NT1 = DHID >> 4;
  if (wave < NT1) {
    const int n0 = wave << 4;
    FragC c;
#pragma unroll
    for (int i = 0; i < 8; i++) c.f[i] = 0.f;
#pragma unroll
    for (int k0 = 0; k0 < DIN; k0 += 32) {
      FragAB a, b;
      loadA(a, ab, DIN, k0, lane);
      loadB(b, w0pk, NT1, k0, n0, lane);
      c.v = __builtin_amdgcn_wmma_f32_16x16x32_bf16(false, a.v, false, b.v,
                                                    (short)0, c.v, false, false);
    }
    const int half = lane >> 4, n = lane & 15;
    const float bs = b0[n0 + n];
#pragma unroll
    for (int i = 0; i < 8; i++)
      hs[(i + 8 * half) * DHID + n0 + n] = fmaxf(c.f[i] + bs, 0.f);
  }
  __syncthreads();

  // ---- LN1 -> bf16 hb
  {
    int r = tid >> 4, sub = tid & 15;
    float s = 0.f, s2 = 0.f;
    for (int d = sub; d < DHID; d += 16) { float w = hs[r * DHID + d]; s += w; s2 += w * w; }
    red[r][sub] = s; red2[r][sub] = s2;
    __syncthreads();
    if (sub == 0) {
      float ts = 0.f, ts2 = 0.f;
#pragma unroll
      for (int j = 0; j < 16; j++) { ts += red[r][j]; ts2 += red2[r][j]; }
      float m = ts / DHID;
      mrow[r] = m;
      rrow[r] = rsqrtf(ts2 / DHID - m * m + 1e-5f);
    }
    __syncthreads();
    for (int idx = tid; idx < 16 * DHID; idx += 256) {
      int r2 = idx / DHID, d = idx - r2 * DHID;
      hb[idx] = f2bf((hs[idx] - mrow[r2]) * rrow[r2] * ln1g[d] + ln1b[d]);
    }
  }
  __syncthreads();

  // ---- GEMM2 + epilogue
  constexpr int NT2 = DOUTPAD >> 4;
  if (wave < NT2) {
    const int n0 = wave << 4;
    FragC c;
#pragma unroll
    for (int i = 0; i < 8; i++) c.f[i] = 0.f;
#pragma unroll
    for (int k0 = 0; k0 < DHID; k0 += 32) {
      FragAB a, b;
      loadA(a, hb, DHID, k0, lane);
      loadB(b, w1pk, NT2, k0, n0, lane);
      c.v = __builtin_amdgcn_wmma_f32_16x16x32_bf16(false, a.v, false, b.v,
                                                    (short)0, c.v, false, false);
    }
    const int half = lane >> 4, n = lane & 15;
    const int cn = n0 + n;
    const float bs = (cn < DOUT) ? b1[cn] : 0.f;
#pragma unroll
    for (int i = 0; i < 8; i++) {
      const int r = i + 8 * half;
      const size_t rg = (size_t)row0 + r;
      float o = c.f[i] + bs;
      if constexpr (MODE == 0) {
        outp[rg * 128 + cn] = o;
      } else if constexpr (MODE == 1) {
        float bl = beta * o + (1.f - beta) * blendSrc[(size_t)ridxE[r] * 128 + cn];
        bl *= rowW[rg];
        atomicAdd(&outp[(size_t)ridxV[r] * 128 + cn], bl);
      } else if constexpr (MODE == 2) {
        float nodev = blendSrc[rg * 128 + cn];
        outp[rg * 128 + cn] = fmaxf(beta * o + (1.f - beta) * nodev, 0.f);
      } else {
        if (cn < DOUT) outp[rg * (size_t)DOUT + cn] = o;
      }
    }
  }
}

// ---------------------------------------------------------------- host driver

struct MlpPtrs { const float *ln0g, *ln0b, *w0, *b0, *ln1g, *ln1b, *w1, *b1; };

extern "C" void kernel_launch(void* const* d_in, const int* in_sizes, int n_in,
                              void* d_out, int out_size, void* d_ws, size_t ws_size,
                              hipStream_t stream) {
  (void)in_sizes; (void)n_in; (void)out_size; (void)ws_size;

  // ---- inputs, flattened in setup_inputs() insertion order
  const float* v_in        = (const float*)d_in[0];
  // d_in[1] = e (unused by reference)
  const int*   vidx        = (const int*)d_in[2];
  const int*   eidx        = (const int*)d_in[3];
  const float* n_weight    = (const float*)d_in[4];
  const float* e_weight    = (const float*)d_in[5];
  const float* n_reg_w     = (const float*)d_in[6];
  const float* e_reg_w     = (const float*)d_in[7];
  const float* n_reg_sum   = (const float*)d_in[8];
  const float* e_reg_sum   = (const float*)d_in[9];
  const float* node_map_w  = (const float*)d_in[10];
  const float* node_map_b  = (const float*)d_in[11];
  const float* edge_map_w  = (const float*)d_in[12];
  const float* edge_map_b  = (const float*)d_in[13];

  auto getM = [&](int base) {
    MlpPtrs m;
    m.ln0g = (const float*)d_in[base + 0];
    m.ln0b = (const float*)d_in[base + 1];
    m.w0   = (const float*)d_in[base + 2];
    m.b0   = (const float*)d_in[base + 3];
    m.ln1g = (const float*)d_in[base + 4];
    m.ln1b = (const float*)d_in[base + 5];
    m.w1   = (const float*)d_in[base + 6];
    m.b1   = (const float*)d_in[base + 7];
    return m;
  };
  MlpPtrs cls    = getM(14);
  MlpPtrs n2m[2] = { getM(22), getM(46) };
  MlpPtrs e2m[2] = { getM(30), getM(54) };
  MlpPtrs nag[2] = { getM(38), getM(62) };

  // ---- workspace layout (f32)
  float* ws = (float*)d_ws;
  size_t o = 0;
  float* v0   = ws + o; o += (size_t)NV_ * 128;
  float* vvb  = ws + o; o += (size_t)NV_ * 128;
  float* e0   = ws + o; o += (size_t)NE_ * 128;
  float* eeb  = ws + o; o += (size_t)NE_ * 128;
  float* tmpV = ws + o; o += (size_t)NV_ * 128;   // nm / node
  float* tmpE = ws + o; o += (size_t)NE_ * 128;   // edge accumulator / edge
  float* nacc = ws + o; o += (size_t)NV_ * 128;
  float* cntE = ws + o; o += NE_;
  float* cntV = ws + o; o += NV_;
  unsigned short* pk = (unsigned short*)(ws + o);
  auto take = [&](size_t n) { unsigned short* p = pk; pk += n; return p; };
  unsigned short* pk_nodemap = take(128 * 128);
  unsigned short* pk_edgemap = take(128 * 128);
  unsigned short *pk_n2m_w0[2], *pk_n2m_w1[2], *pk_e2m_w0[2], *pk_e2m_w1[2],
                 *pk_nag_w0[2], *pk_nag_w1[2];
  for (int i = 0; i < 2; i++) {
    pk_n2m_w0[i] = take(128 * 128); pk_n2m_w1[i] = take(128 * 128);
    pk_e2m_w0[i] = take(256 * 128); pk_e2m_w1[i] = take(128 * 128);
    pk_nag_w0[i] = take(128 * 128); pk_nag_w1[i] = take(128 * 128);
  }
  unsigned short* pk_cls_w0 = take(128 * 64);
  unsigned short* pk_cls_w1 = take(64 * 48);

  const dim3 B(256);
  auto nb = [](long long n) { return dim3((unsigned)((n + 255) / 256)); };

  // ---- counts
  fill_zero_k<<<nb(NE_), B, 0, stream>>>(cntE, NE_);
  fill_zero_k<<<nb(NV_), B, 0, stream>>>(cntV, NV_);
  fill_zero_k<<<nb((long long)NE_ * 128), B, 0, stream>>>(tmpE, (long long)NE_ * 128);
  seg_count_k<<<nb(NNZ_), B, 0, stream>>>(eidx, cntE, NNZ_);
  seg_count_k<<<nb(NNZ_), B, 0, stream>>>(vidx, cntV, NNZ_);

  // ---- pack all GEMM weights to bf16 WMMA layout
  auto pack = [&](const float* w, unsigned short* dst, int K, int N, int Np) {
    int total = (K >> 5) * (Np >> 4) * 512;
    pack_w_bf16_k<<<nb(total), B, 0, stream>>>(w, dst, K, N, Np);
  };
  pack(node_map_w, pk_nodemap, 128, 128, 128);
  pack(edge_map_w, pk_edgemap, 128, 128, 128);
  for (int i = 0; i < 2; i++) {
    pack(n2m[i].w0, pk_n2m_w0[i], 128, 128, 128);
    pack(n2m[i].w1, pk_n2m_w1[i], 128, 128, 128);
    pack(e2m[i].w0, pk_e2m_w0[i], 256, 128, 128);
    pack(e2m[i].w1, pk_e2m_w1[i], 128, 128, 128);
    pack(nag[i].w0, pk_nag_w0[i], 128, 128, 128);
    pack(nag[i].w1, pk_nag_w1[i], 128, 128, 128);
  }
  pack(cls.w0, pk_cls_w0, 128, 64, 64);
  pack(cls.w1, pk_cls_w1, 64, 40, 48);

  // ---- initial embeddings
  gather_scatter_add_k<<<nb((long long)NNZ_ * 32), B, 0, stream>>>(
      v_in, vidx, eidx, nullptr, tmpE, NNZ_);
  div_count_k<<<nb((long long)NE_ * 128), B, 0, stream>>>(tmpE, cntE, (long long)NE_ * 128);
  linear_relu_k<<<NE_ / 16, B, 0, stream>>>(tmpE, pk_edgemap, edge_map_b, e0);
  linear_relu_k<<<NV_ / 16, B, 0, stream>>>(v_in, pk_nodemap, node_map_b, v0);
  hipMemcpyAsync(vvb, v0, (size_t)NV_ * 128 * sizeof(float), hipMemcpyDeviceToDevice, stream);
  hipMemcpyAsync(eeb, e0, (size_t)NE_ * 128 * sizeof(float), hipMemcpyDeviceToDevice, stream);

  // ---- layers
  for (int i = 0; i < 2; i++) {
    const float beta = logf(0.5f / (float)(i + 1) + 1.f);

    // nm = MLP(vv * n_weight)
    mlp_fused_k<0, 128, 128, 128, 128><<<NV_ / 16, B, 0, stream>>>(
        vvb, nullptr, nullptr, nullptr, n_weight,
        n2m[i].ln0g, n2m[i].ln0b, pk_n2m_w0[i], n2m[i].b0,
        n2m[i].ln1g, n2m[i].ln1b, pk_n2m_w1[i], n2m[i].b1,
        0.f, nullptr, nullptr, tmpV);

    // edge = scatter_mean(nm[vidx]*n_reg_w, eidx); blend with e0; /e_reg_sum
    fill_zero_k<<<nb((long long)NE_ * 128), B, 0, stream>>>(tmpE, (long long)NE_ * 128);
    gather_scatter_add_k<<<nb((long long)NNZ_ * 32), B, 0, stream>>>(
        tmpV, vidx, eidx, n_reg_w, tmpE, NNZ_);
    edge_finalize_k<<<nb((long long)NE_ * 128), B, 0, stream>>>(
        tmpE, cntE, e0, e_reg_sum, (long long)NE_ * 128);

    // edge_msg MLP over NNZ rows, fused gather/concat + blend + scatter-add
    fill_zero_k<<<nb((long long)NV_ * 128), B, 0, stream>>>(nacc, (long long)NV_ * 128);
    mlp_fused_k<1, 256, 128, 128, 128><<<NNZ_ / 16, B, 0, stream>>>(
        vvb, tmpE, vidx, eidx, e_weight,
        e2m[i].ln0g, e2m[i].ln0b, pk_e2m_w0[i], e2m[i].b0,
        e2m[i].ln1g, e2m[i].ln1b, pk_e2m_w1[i], e2m[i].b1,
        beta, eeb, e_reg_w, nacc);

    // node = (1-a)*mean/n_reg_sum + a*v0 ; vv = relu(beta*MLP(node)+(1-beta)*node)
    node_finalize_k<<<nb((long long)NV_ * 128), B, 0, stream>>>(
        nacc, cntV, n_reg_sum, v0, tmpV, (long long)NV_ * 128);
    mlp_fused_k<2, 128, 128, 128, 128><<<NV_ / 16, B, 0, stream>>>(
        tmpV, nullptr, nullptr, nullptr, nullptr,
        nag[i].ln0g, nag[i].ln0b, pk_nag_w0[i], nag[i].b0,
        nag[i].ln1g, nag[i].ln1b, pk_nag_w1[i], nag[i].b1,
        beta, tmpV, nullptr, vvb);

    relu_copy_k<<<nb((long long)NE_ * 128), B, 0, stream>>>(eeb, tmpE, (long long)NE_ * 128);
  }

  // ---- outputs: [vv | ee | pred]
  float* outv = (float*)d_out;
  float* oute = outv + (size_t)NV_ * 128;
  float* outp = oute + (size_t)NE_ * 128;
  mlp_fused_k<3, 128, 64, 48, NCLS_><<<NV_ / 16, B, 0, stream>>>(
      vvb, nullptr, nullptr, nullptr, nullptr,
      nullptr, nullptr, pk_cls_w0, cls.b0,
      cls.ln1g, cls.ln1b, pk_cls_w1, cls.b1,
      0.f, nullptr, nullptr, outp);
  hipMemcpyAsync(outv, vvb, (size_t)NV_ * 128 * sizeof(float), hipMemcpyDeviceToDevice, stream);
  hipMemcpyAsync(oute, eeb, (size_t)NE_ * 128 * sizeof(float), hipMemcpyDeviceToDevice, stream);
}
